// GCN_5050881540192
// MI455X (gfx1250) — compile-verified, matrix operand order checked by reading, not run on hardware
//
#include <hip/hip_runtime.h>
#include <math.h>

typedef __bf16 v16bf __attribute__((ext_vector_type(16)));
typedef __bf16 bf8   __attribute__((ext_vector_type(8)));
typedef float  v8f   __attribute__((ext_vector_type(8)));

#define BN_EPS 1e-5f

// ---------------- utility ----------------
__global__ void k_zero_f32(float* p, size_t n) {
  size_t i = (size_t)blockIdx.x * blockDim.x + threadIdx.x;
  if (i < n) p[i] = 0.0f;
}

__global__ void k_degree(const long long* __restrict__ dst, float* __restrict__ deg, int E) {
  int e = blockIdx.x * blockDim.x + threadIdx.x;
  if (e < E) atomicAdd(&deg[(int)dst[e]], 1.0f);
}

// deg -> rsqrt(deg + 1)  (self-loop folded in; deg+1 >= 1 always)
__global__ void k_dinv(float* __restrict__ d, int n) {
  int i = blockIdx.x * blockDim.x + threadIdx.x;
  if (i < n) d[i] = rsqrtf(d[i] + 1.0f);
}

__global__ void k_f32_to_bf16(const float* __restrict__ in, __bf16* __restrict__ out, size_t n) {
  size_t i = (size_t)blockIdx.x * blockDim.x + threadIdx.x;
  if (i < n) out[i] = (__bf16)in[i];
}

// ---------------- weight pre-pack into WMMA B-fragment lane layout ----------------
// Bpack element index = (((t*ksteps + ks)*32 + lane)*16 + j)
//   lane: khalf = lane>>4, col = t*16 + (lane&15)
//   j<8 : K = ks*32 + khalf*8 + j ; j>=8 : K = ks*32 + 16 + khalf*8 + (j-8)
// Columns >= M are zero-padded.
__global__ void k_pack_w(const float* __restrict__ W, __bf16* __restrict__ Bp,
                         int K, int M, int tilesP) {
  int ksteps = K >> 5;
  size_t total = (size_t)tilesP * ksteps * 512;
  size_t idx = (size_t)blockIdx.x * blockDim.x + threadIdx.x;
  if (idx >= total) return;
  int j    = (int)(idx & 15);
  int lane = (int)((idx >> 4) & 31);
  int comb = (int)(idx >> 9);
  int ks   = comb % ksteps;
  int t    = comb / ksteps;
  int khalf = lane >> 4;
  int col   = t * 16 + (lane & 15);
  int k     = ks * 32 + khalf * 8 + (j & 7) + ((j >> 3) * 16);
  float v = (col < M) ? W[(size_t)k * M + col] : 0.0f;
  Bp[idx] = (__bf16)v;
}

// ---------------- dense transform: D(Npad x M) = A(Npad x K bf16) * Bpack ----------------
// One wave32 per (16-row tile) x (4 column tiles). A fragment loaded once per K-step
// (2x global_load_b128), reused across 4 WMMAs. B fragments are pre-packed ->
// 2x coalesced global_load_b128 each. K compile-time -> fully unrolled.
template<int K>
__global__ void k_wmma_gemm4(const __bf16* __restrict__ A, const __bf16* __restrict__ Bp,
                             float* __restrict__ D, int M) {
  constexpr int KSTEPS = K / 32;
  const int lane  = threadIdx.x;           // 0..31
  const int m16   = lane & 15;
  const int khalf = lane >> 4;
  const size_t rowBase = (size_t)blockIdx.x * 16;
  const int tile0 = blockIdx.y * 4;

  const bf8* arow  = (const bf8*)(A + (rowBase + m16) * K);
  const bf8* bbase = (const bf8*)Bp;

  v8f acc[4] = {v8f{}, v8f{}, v8f{}, v8f{}};

  #pragma unroll
  for (int ks = 0; ks < KSTEPS; ++ks) {
    bf8 alo = arow[ks * 4 + khalf];
    bf8 ahi = arow[ks * 4 + 2 + khalf];
    v16bf av = __builtin_shufflevector(alo, ahi, 0,1,2,3,4,5,6,7,8,9,10,11,12,13,14,15);
    #pragma unroll
    for (int t = 0; t < 4; ++t) {
      size_t b = (((size_t)(tile0 + t) * KSTEPS + ks) * 32 + lane) * 2; // bf8 units
      bf8 blo = bbase[b];
      bf8 bhi = bbase[b + 1];
      v16bf bv = __builtin_shufflevector(blo, bhi, 0,1,2,3,4,5,6,7,8,9,10,11,12,13,14,15);
      acc[t] = __builtin_amdgcn_wmma_f32_16x16x32_bf16(
          false, av, false, bv, (short)0, acc[t], false, false);
    }
  }

  // D layout: lane (l&15) = column; VGPR r -> row r + 8*(l>>4). Rows always < Npad.
  #pragma unroll
  for (int t = 0; t < 4; ++t) {
    int nCol = (tile0 + t) * 16 + m16;
    if (nCol < M) {
      #pragma unroll
      for (int r = 0; r < 8; ++r) {
        D[(rowBase + r + 8 * khalf) * M + nCol] = acc[t][r];
      }
    }
  }
}

// ---------------- sparse aggregation ----------------
// Hidden layers (C multiple of 4): one block of C/4 threads per edge;
// float4 gather + 4 f32 atomics per thread (atomics resolve in 192MB L2).
__global__ void k_spmm_v4(const long long* __restrict__ src, const long long* __restrict__ dst,
                          const float* __restrict__ dinv, const float* __restrict__ hin,
                          float* __restrict__ hout, int C, int E) {
  int e = blockIdx.x;
  if (e >= E) return;
  int s = (int)src[e];
  int d = (int)dst[e];
  float w = dinv[s] * dinv[d];
  int c4 = threadIdx.x * 4;
  const float4* hs = (const float4*)(hin + (size_t)s * C + c4);
  float* hd = hout + (size_t)d * C + c4;
  float4 v = *hs;
  atomicAdd(hd + 0, v.x * w);
  atomicAdd(hd + 1, v.y * w);
  atomicAdd(hd + 2, v.z * w);
  atomicAdd(hd + 3, v.w * w);
}

// Generic variant for C=40.
__global__ void k_spmm(const long long* __restrict__ src, const long long* __restrict__ dst,
                       const float* __restrict__ dinv, const float* __restrict__ hin,
                       float* __restrict__ hout, int C, int E) {
  int e = blockIdx.x;
  if (e >= E) return;
  int s = (int)src[e];
  int d = (int)dst[e];
  float w = dinv[s] * dinv[d];
  const float* hs = hin + (size_t)s * C;
  float* hd = hout + (size_t)d * C;
  for (int c = threadIdx.x; c < C; c += blockDim.x)
    atomicAdd(&hd[c], hs[c] * w);
}

// self-loop term (weight dinv[i]^2) + bias, non-atomic (runs after spmm)
__global__ void k_selfloop_bias(const float* __restrict__ hlin, const float* __restrict__ dinv,
                                const float* __restrict__ b, float* __restrict__ hagg,
                                int Nn, int C) {
  size_t idx = (size_t)blockIdx.x * blockDim.x + threadIdx.x;
  if (idx >= (size_t)Nn * C) return;
  int i = (int)(idx / C);
  int c = (int)(idx % C);
  float dv = dinv[i];
  hagg[idx] = hagg[idx] + hlin[idx] * dv * dv + b[c];
}

// ---------------- batch norm ----------------
__global__ void k_bn_stats(const float* __restrict__ h, float* __restrict__ sums,
                           int Nn, int C) {
  int c = threadIdx.x;             // blockDim.x == C
  int r0 = blockIdx.x * 256;
  int r1 = r0 + 256; if (r1 > Nn) r1 = Nn;
  float s = 0.0f, ss = 0.0f;
  for (int r = r0; r < r1; ++r) {
    float v = h[(size_t)r * C + c];
    s += v; ss += v * v;
  }
  atomicAdd(&sums[c], s);
  atomicAdd(&sums[C + c], ss);
}

__global__ void k_bn_relu_bf16(const float* __restrict__ h, const float* __restrict__ sums,
                               const float* __restrict__ gamma, const float* __restrict__ beta,
                               __bf16* __restrict__ out, int Nn, int C) {
  size_t idx = (size_t)blockIdx.x * blockDim.x + threadIdx.x;
  if (idx >= (size_t)Nn * C) return;
  int c = (int)(idx % C);
  float invN = 1.0f / (float)Nn;
  float mean = sums[c] * invN;
  float var  = sums[C + c] * invN - mean * mean;
  float y = (h[idx] - mean) * rsqrtf(var + BN_EPS) * gamma[c] + beta[c];
  out[idx] = (__bf16)fmaxf(y, 0.0f);
}

// ---------------- final log_softmax over 40 classes (bias already added) ----------------
__global__ void k_log_softmax(const float* __restrict__ h, float* __restrict__ out,
                              int Nn, int C) {
  int i = blockIdx.x * blockDim.x + threadIdx.x;
  if (i >= Nn) return;
  const float* r = h + (size_t)i * C;
  float mx = -1e30f;
  for (int c = 0; c < C; ++c) mx = fmaxf(mx, r[c]);
  float s = 0.0f;
  for (int c = 0; c < C; ++c) s += __expf(r[c] - mx);
  float lse = mx + __logf(s);
  float* o = out + (size_t)i * C;
  for (int c = 0; c < C; ++c) o[c] = r[c] - lse;
}

// ---------------- host orchestration ----------------
extern "C" void kernel_launch(void* const* d_in, const int* in_sizes, int n_in,
                              void* d_out, int out_size, void* d_ws, size_t ws_size,
                              hipStream_t stream) {
  (void)n_in; (void)out_size; (void)ws_size;

  const float*     x   = (const float*)d_in[0];
  const long long* ei  = (const long long*)d_in[1];   // int64 edge_index [2, E]
  const float*     W1  = (const float*)d_in[2];
  const float*     b1  = (const float*)d_in[3];
  const float*     W2  = (const float*)d_in[4];
  const float*     b2  = (const float*)d_in[5];
  const float*     W3  = (const float*)d_in[6];
  const float*     b3  = (const float*)d_in[7];
  const float*     g1  = (const float*)d_in[8];
  const float*     be1 = (const float*)d_in[9];
  const float*     g2  = (const float*)d_in[10];
  const float*     be2 = (const float*)d_in[11];

  const int IN_C = 128, HID = 256, OUT = 40;
  const int Nn   = in_sizes[0] / IN_C;
  const int E    = in_sizes[1] / 2;
  const int Npad = (Nn + 15) & ~15;
  const long long* src = ei;
  const long long* dst = ei + E;

  // carve workspace (256B aligned)
  char* p = (char*)d_ws;
  auto carve = [&](size_t bytes) -> void* {
    void* r = (void*)p;
    p += (bytes + 255) & ~(size_t)255;
    return r;
  };
  float*  dinv = (float*) carve((size_t)Nn * 4);
  float*  sums = (float*) carve((size_t)2 * HID * 4);
  __bf16* bpk  = (__bf16*)carve((size_t)16 * 8 * 512 * 2);      // packed weights (<=128KB)
  __bf16* hbf  = (__bf16*)carve((size_t)Npad * HID * 2);
  float*  hlin = (float*) carve((size_t)Npad * HID * 4);
  float*  hagg = (float*) carve((size_t)Npad * HID * 4);

  auto blocks = [](size_t n, int t) { return (unsigned)((n + (size_t)t - 1) / (size_t)t); };

  // ---- symmetric normalization ----
  k_zero_f32<<<blocks((size_t)Nn, 256), 256, 0, stream>>>(dinv, (size_t)Nn);
  k_degree  <<<blocks((size_t)E, 256), 256, 0, stream>>>(dst, dinv, E);
  k_dinv    <<<blocks((size_t)Nn, 256), 256, 0, stream>>>(dinv, Nn);

  // zero hbf once so padded rows contribute zeros to WMMA
  k_zero_f32<<<blocks((size_t)Npad * HID / 2, 256), 256, 0, stream>>>((float*)hbf, (size_t)Npad * HID / 2);

  // ---- layer 1: x@W1 -> spmm -> +self+b1 -> BN -> ReLU -> bf16 ----
  k_f32_to_bf16<<<blocks((size_t)Nn * IN_C, 256), 256, 0, stream>>>(x, hbf, (size_t)Nn * IN_C);
  {
    const int tilesP = 16;                     // HID/16
    size_t total = (size_t)tilesP * (IN_C / 32) * 512;
    k_pack_w<<<blocks(total, 256), 256, 0, stream>>>(W1, bpk, IN_C, HID, tilesP);
    dim3 grid(Npad / 16, tilesP / 4);
    k_wmma_gemm4<128><<<grid, 32, 0, stream>>>(hbf, bpk, hlin, HID);
  }
  k_zero_f32<<<blocks((size_t)Nn * HID, 256), 256, 0, stream>>>(hagg, (size_t)Nn * HID);
  k_spmm_v4<<<(unsigned)E, HID / 4, 0, stream>>>(src, dst, dinv, hlin, hagg, HID, E);
  k_selfloop_bias<<<blocks((size_t)Nn * HID, 256), 256, 0, stream>>>(hlin, dinv, b1, hagg, Nn, HID);
  k_zero_f32<<<2, 256, 0, stream>>>(sums, (size_t)2 * HID);
  k_bn_stats<<<blocks((size_t)Nn, 256), HID, 0, stream>>>(hagg, sums, Nn, HID);
  k_bn_relu_bf16<<<blocks((size_t)Nn * HID, 256), 256, 0, stream>>>(hagg, sums, g1, be1, hbf, Nn, HID);

  // ---- layer 2 ----
  {
    const int tilesP = 16;
    size_t total = (size_t)tilesP * (HID / 32) * 512;
    k_pack_w<<<blocks(total, 256), 256, 0, stream>>>(W2, bpk, HID, HID, tilesP);
    dim3 grid(Npad / 16, tilesP / 4);
    k_wmma_gemm4<256><<<grid, 32, 0, stream>>>(hbf, bpk, hlin, HID);
  }
  k_zero_f32<<<blocks((size_t)Nn * HID, 256), 256, 0, stream>>>(hagg, (size_t)Nn * HID);
  k_spmm_v4<<<(unsigned)E, HID / 4, 0, stream>>>(src, dst, dinv, hlin, hagg, HID, E);
  k_selfloop_bias<<<blocks((size_t)Nn * HID, 256), 256, 0, stream>>>(hlin, dinv, b2, hagg, Nn, HID);
  k_zero_f32<<<2, 256, 0, stream>>>(sums, (size_t)2 * HID);
  k_bn_stats<<<blocks((size_t)Nn, 256), HID, 0, stream>>>(hagg, sums, Nn, HID);
  k_bn_relu_bf16<<<blocks((size_t)Nn * HID, 256), 256, 0, stream>>>(hagg, sums, g2, be2, hbf, Nn, HID);

  // ---- layer 3: h@W3 (M=40, tiles padded to 4) -> spmm -> +self+b3 -> log_softmax ----
  {
    const int tilesP = 4;                      // ceil(40/16)=3, padded to 4
    size_t total = (size_t)tilesP * (HID / 32) * 512;
    k_pack_w<<<blocks(total, 256), 256, 0, stream>>>(W3, bpk, HID, OUT, tilesP);
    dim3 grid(Npad / 16, tilesP / 4);
    k_wmma_gemm4<256><<<grid, 32, 0, stream>>>(hbf, bpk, hlin, OUT);
  }
  k_zero_f32<<<blocks((size_t)Nn * OUT, 256), 256, 0, stream>>>(hagg, (size_t)Nn * OUT);
  k_spmm<<<(unsigned)E, 64, 0, stream>>>(src, dst, dinv, hlin, hagg, OUT, E);
  k_selfloop_bias<<<blocks((size_t)Nn * OUT, 256), 256, 0, stream>>>(hlin, dinv, b3, hagg, Nn, OUT);
  k_log_softmax<<<blocks((size_t)Nn, 128), 128, 0, stream>>>(hagg, (float*)d_out, Nn, OUT);
}
